// AltInnerProductDecoder_87789131530252
// MI455X (gfx1250) — compile-verified, matrix operand order checked by reading, not run on hardware
//
#include <hip/hip_runtime.h>

#define NN     2048
#define NLAT   64
#define NHID   32
#define LN_EPS 1e-5f
#define JBLK   512                 // columns per block: 8 waves x 4 tiles x 16

typedef __attribute__((ext_vector_type(16))) _Float16 v16h;
typedef __attribute__((ext_vector_type(8)))  float    v8f;
typedef __attribute__((ext_vector_type(4)))  float    v4f;

union F8  { v8f  v; float    f[8];  };
union H16 { v16h v; _Float16 h[16]; };

// sum across each 16-lane row with DPP16 only (no LDS pipe):
// ror:8, ror:4 then quad_perm swaps complete the 16-lane rotation schedule.
__device__ __forceinline__ float row16_sum(float x) {
  int t;
  t = __builtin_amdgcn_update_dpp(0, __float_as_int(x), 0x128, 0xF, 0xF, true); // row_ror:8
  x += __int_as_float(t);
  t = __builtin_amdgcn_update_dpp(0, __float_as_int(x), 0x124, 0xF, 0xF, true); // row_ror:4
  x += __int_as_float(t);
  t = __builtin_amdgcn_update_dpp(0, __float_as_int(x), 0x04E, 0xF, 0xF, true); // qp [2,3,0,1]
  x += __int_as_float(t);
  t = __builtin_amdgcn_update_dpp(0, __float_as_int(x), 0x0B1, 0xF, 0xF, true); // qp [1,0,3,2]
  x += __int_as_float(t);
  return x;
}

// exchange with lane^16 via ds_swizzle SWAPX16 (xor_mask=0x10, and_mask=0x1f)
__device__ __forceinline__ float swap16(float x) {
  return __int_as_float(__builtin_amdgcn_ds_swizzle(__float_as_int(x), 0x401F));
}

// single v_max_num_f32 (no canonicalize pair like fmaxf)
__device__ __forceinline__ float relu(float x) { return x > 0.0f ? x : 0.0f; }

__global__ __launch_bounds__(256) void zero_kernel(float* __restrict__ out) {
  int idx = blockIdx.x * blockDim.x + threadIdx.x;
  v4f zv = {0.0f, 0.0f, 0.0f, 0.0f};
  reinterpret_cast<v4f*>(out)[idx] = zv;
}

// One wave32 processes 4 WMMA M-tiles of 16 pairs each (i, j in [jw, jw+64)).
// Lane (q = lane&15, half = lane>>4) stages pair m=q of each tile; per the
// CDNA5 16-bit A layout it holds features {half*8+t, 16+half*8+t : t=0..7} of
// each 32-wide K chunk. The 65th (distance) feature goes through a third,
// zero-padded WMMA K-chunk so no cross-lane broadcast is ever needed.
// WMMAs alternate acc0/acc1 chains and each a-tile build is interleaved with
// in-flight WMMAs of the other chain to avoid VALU->WMMA hazard NOPs.
__global__ __launch_bounds__(256) void pair_mlp_kernel(
    const float* __restrict__ z,  const float* __restrict__ w1,
    const float* __restrict__ b1, const float* __restrict__ w2,
    const float* __restrict__ b2, const float* __restrict__ gamma,
    const float* __restrict__ beta, float* __restrict__ out)
{
  const int i    = blockIdx.y;
  const int jblk = blockIdx.x * JBLK;
  if (jblk + (JBLK - 1) <= i) return;          // block-uniform skip (lower tri)
  const int lane = threadIdx.x & 31;
  const int q    = lane & 15;
  const int half = lane >> 4;
  const int wv   = __builtin_amdgcn_readfirstlane((int)(threadIdx.x >> 5));
  const int jw   = jblk + wv * 64;             // this wave's 64-column span
  if (jw + 63 <= i) return;                    // wave-uniform skip (scalar)

  // ---- per-wave invariants: z[i] features, gamma/beta, w1 B tiles ----------
  float zi[32], gm[32], bt[32];
  const float* zrow_i = z + i * NLAT;
  #pragma unroll
  for (int g = 0; g < 4; ++g) {
    const int base = g * 16 + half * 8;
    *reinterpret_cast<v4f*>(zi + g*8)     = *reinterpret_cast<const v4f*>(zrow_i + base);
    *reinterpret_cast<v4f*>(zi + g*8 + 4) = *reinterpret_cast<const v4f*>(zrow_i + base + 4);
    *reinterpret_cast<v4f*>(gm + g*8)     = *reinterpret_cast<const v4f*>(gamma + base);
    *reinterpret_cast<v4f*>(gm + g*8 + 4) = *reinterpret_cast<const v4f*>(gamma + base + 4);
    *reinterpret_cast<v4f*>(bt + g*8)     = *reinterpret_cast<const v4f*>(beta + base);
    *reinterpret_cast<v4f*>(bt + g*8 + 4) = *reinterpret_cast<const v4f*>(beta + base + 4);
  }

  H16 bmat[2][2];                              // [K-chunk][N-tile] of w1 (f16)
  #pragma unroll
  for (int c = 0; c < 2; ++c)
    #pragma unroll
    for (int nt = 0; nt < 2; ++nt)
      #pragma unroll
      for (int t = 0; t < 16; ++t) {
        int k = c*32 + ((t < 8) ? (half*8 + t) : (16 + half*8 + (t - 8)));
        bmat[c][nt].h[t] = (_Float16)w1[k * NHID + nt*16 + q];
      }

  // distance-row B tiles: only K=0 (lanes 0-15, slot 0) carries w1[64][n]
  H16 bdist[2];
  #pragma unroll
  for (int nt = 0; nt < 2; ++nt) {
    bdist[nt].v = (v16h){};
    bdist[nt].h[0] = (half == 0) ? (_Float16)w1[64*NHID + nt*16 + q] : (_Float16)0.0f;
  }

  const float b1n0 = b1[q],      b1n1 = b1[16 + q];
  const float w2n0 = w2[q],      w2n1 = w2[16 + q];
  const float g64  = gamma[64],  be64 = beta[64];
  const float b2v  = b2[0];

  // ---- loop over 4 j-tiles of 16 pairs each --------------------------------
  #pragma unroll
  for (int t = 0; t < 4; ++t) {
    const int j0 = jw + t * 16;
    if (j0 + 15 <= i) continue;                // uniform: whole tile lower-tri
    const int j = j0 + q;                      // column staged by this lane
    if (t < 3)                                 // global_prefetch_b8: next tile
      __builtin_prefetch(z + (j + 16) * NLAT, 0, 3);

    // zp = z[i]*z[j] for this lane's 32 of 64 features + LN statistics
    float zp[32];
    const float* zrow_j = z + j * NLAT;
    #pragma unroll
    for (int g = 0; g < 4; ++g) {
      const int base = g * 16 + half * 8;
      *reinterpret_cast<v4f*>(zp + g*8)     = *reinterpret_cast<const v4f*>(zrow_j + base);
      *reinterpret_cast<v4f*>(zp + g*8 + 4) = *reinterpret_cast<const v4f*>(zrow_j + base + 4);
    }
    float s = 0.0f, s2 = 0.0f;
    #pragma unroll
    for (int u = 0; u < 32; ++u) {
      zp[u] *= zi[u];
      s  += zp[u];
      s2 += zp[u] * zp[u];
    }
    s  += swap16(s);                           // lanes q / q+16 share pair q
    s2 += swap16(s2);
    const float d    = (float)(j - i);
    const float mean = (s + d) * (1.0f / 65.0f);
    const float var  = (s2 + d * d) * (1.0f / 65.0f) - mean * mean;
    const float rstd = rsqrtf(var + LN_EPS);

    // distance-feature A tile first (only needs d65): K=0 on lanes 0-15
    const float d65 = (d - mean) * rstd * g64 + be64;
    H16 adist;
    adist.v = (v16h){};
    adist.h[0] = (half == 0) ? (_Float16)d65 : (_Float16)0.0f;

    // C accumulators pre-loaded with b1 (C layout: N = lane%16 on all VGPRs)
    F8 acc0, acc1;
    #pragma unroll
    for (int v = 0; v < 8; ++v) { acc0.f[v] = b1n0; acc1.f[v] = b1n1; }

    // dist WMMAs first: their A operand is ready earliest; alternate chains
    acc0.v = __builtin_amdgcn_wmma_f32_16x16x32_f16(false, adist.v, false, bdist[0].v,
                                                    (short)0, acc0.v, false, false);
    acc1.v = __builtin_amdgcn_wmma_f32_16x16x32_f16(false, adist.v, false, bdist[1].v,
                                                    (short)0, acc1.v, false, false);

    // build a[0] (features 0..31) while dist WMMAs are in flight
    H16 a0;
    #pragma unroll
    for (int u = 0; u < 16; ++u)
      a0.h[u] = (_Float16)((zp[u] - mean) * rstd * gm[u] + bt[u]);
    acc0.v = __builtin_amdgcn_wmma_f32_16x16x32_f16(false, a0.v, false, bmat[0][0].v,
                                                    (short)0, acc0.v, false, false);
    acc1.v = __builtin_amdgcn_wmma_f32_16x16x32_f16(false, a0.v, false, bmat[0][1].v,
                                                    (short)0, acc1.v, false, false);

    // build a[1] (features 32..63) while a0 WMMAs are in flight
    H16 a1;
    #pragma unroll
    for (int u = 0; u < 16; ++u)
      a1.h[u] = (_Float16)((zp[16 + u] - mean) * rstd * gm[16 + u] + bt[16 + u]);
    acc0.v = __builtin_amdgcn_wmma_f32_16x16x32_f16(false, a1.v, false, bmat[1][0].v,
                                                    (short)0, acc0.v, false, false);
    acc1.v = __builtin_amdgcn_wmma_f32_16x16x32_f16(false, a1.v, false, bmat[1][1].v,
                                                    (short)0, acc1.v, false, false);

    // ReLU + second linear (32->1): DPP16 row sum over hidden dim
    float r[8];
    #pragma unroll
    for (int v = 0; v < 8; ++v) {
      float rv = relu(acc0.f[v]) * w2n0 + relu(acc1.f[v]) * w2n1;
      r[v] = row16_sum(rv) + b2v;
    }

    // scatter: lanes q=0..7 of each half write pair m = q + 8*half
    float val = r[0];
    #pragma unroll
    for (int v = 1; v < 8; ++v) if (q == v) val = r[v];
    if (q < 8) {
      int m  = q + 8 * half;
      int jj = j0 + m;
      if (jj > i) out[i * NN + jj] = val;
    }
  }
}

extern "C" void kernel_launch(void* const* d_in, const int* in_sizes, int n_in,
                              void* d_out, int out_size, void* d_ws, size_t ws_size,
                              hipStream_t stream) {
  const float* z     = (const float*)d_in[0];
  const float* w1    = (const float*)d_in[1];
  const float* b1    = (const float*)d_in[2];
  const float* w2    = (const float*)d_in[3];
  const float* b2    = (const float*)d_in[4];
  const float* gamma = (const float*)d_in[5];
  const float* beta  = (const float*)d_in[6];
  float* out = (float*)d_out;

  // clear the 16 MB adjacency (harness poisons d_out), then fill upper tri
  const int nvec = (NN * NN) / 4;                       // float4 stores
  zero_kernel<<<nvec / 256, 256, 0, stream>>>(out);

  dim3 grid(NN / JBLK, NN);                             // 512 columns per block
  pair_mlp_kernel<<<grid, 256, 0, stream>>>(z, w1, b1, w2, b2, gamma, beta, out);
}
